// LocalSelfAttentionBase_36258113913321
// MI455X (gfx1250) — matrix operand order repaired and checked by reading, not compiled, not run on hardware
//
#include <hip/hip_runtime.h>

// ---------------------------------------------------------------------------
// Sparse-conv kernel-map construction for MI455X (gfx1250, wave32).
//
// Strategy: key space is 8 * 132^3 = 18,399,744 < 2^25, so unique-rank is
// computed with a presence bitmap (2.3 MB, L2-resident on the 192 MB L2) and
// a hierarchical prefix-popcount instead of a 6.2M-element sort. The per-wave
// 256-element prefix sum is done with two v_wmma_f32_16x16x32_bf16 ops
// (tensor-core scan: X*U for row scans, Lstrict*X + shuffle-reduce for row
// offsets). All values are small integers, exact in bf16 (<=32) and the f32
// accumulator (<2^24).
// ---------------------------------------------------------------------------

typedef __attribute__((ext_vector_type(16))) __bf16 v16bf;
typedef __attribute__((ext_vector_type(8)))  float  v8f;

#define GRID_R   128
#define PAD      2
#define BASE     (GRID_R + 2 * PAD)              // 132
#define NBATCH   8
#define KEYSPACE (NBATCH * BASE * BASE * BASE)   // 18,399,744 < 2^25
#define NWORDS   ((KEYSPACE + 31) / 32)          // 574,992 bitmap words
#define WPB      2048                            // words per scan block (8 waves x 256)
#define NBLK     ((NWORDS + WPB - 1) / WPB)      // 281 (<512)

// ---------------------------------------------------------------------------
__global__ void zero_bitmap_kernel(unsigned* __restrict__ bm, int n) {
  int i = blockIdx.x * blockDim.x + threadIdx.x;
  if (i < n) bm[i] = 0u;
}

// ---------------------------------------------------------------------------
// Expand coords by kernel offsets; write out_key + static kernel_map columns;
// set presence bit for each encoded key (atomicOr stays in L2).
__global__ void expand_kernel(const int* __restrict__ si,   // [4, N]
                              const int* __restrict__ ko,   // [K, 3]
                              int N, int K, long long NK,
                              int* __restrict__ out,        // kernel_map | out_key
                              unsigned* __restrict__ bitmap)
{
  long long i = (long long)blockIdx.x * blockDim.x + threadIdx.x;
  if (i >= NK) return;
  int n = (int)(i / K);
  int k = (int)(i % K);

  int b = si[(long long)0 * N + n];
  int x = si[(long long)1 * N + n] + ko[3 * k + 0];
  int y = si[(long long)2 * N + n] + ko[3 * k + 1];
  int z = si[(long long)3 * N + n] + ko[3 * k + 2];

  // kernel_map columns 0 and 2 (column 1 = rank, filled by rank_kernel)
  out[3 * i + 0] = n;
  out[3 * i + 2] = k;

  // out_key rows, concatenated after kernel_map
  long long ob = 3 * NK + 4 * i;
  out[ob + 0] = b;
  out[ob + 1] = x;
  out[ob + 2] = y;
  out[ob + 3] = z;

  int key = ((b * BASE + (x + PAD)) * BASE + (y + PAD)) * BASE + (z + PAD);
  atomicOr(&bitmap[key >> 5], 1u << (key & 31));
}

// ---------------------------------------------------------------------------
// Per-block exclusive scan of word popcounts. Each wave scans 256 words with
// a WMMA tensor-core scan; 8 waves per block combine through LDS.
__global__ __launch_bounds__(256) void scan_blocks_kernel(
    const unsigned* __restrict__ bitmap,
    int*            __restrict__ wordScan,    // exclusive-within-block
    unsigned*       __restrict__ blockSums)
{
  __shared__ float s_cnt[WPB];
  __shared__ float s_wsum[8];
  __shared__ float s_woff[8];

  const int tid   = threadIdx.x;
  const int wave  = tid >> 5;
  const int lane  = tid & 31;
  const int lhalf = (lane >= 16) ? 1 : 0;
  const int ln16  = lane & 15;
  const int blockBase = blockIdx.x * WPB;
  const int wbase = wave * 256;

  // Stage popcounts of this block's 2048 words into LDS (coalesced).
#pragma unroll
  for (int s = 0; s < 8; ++s) {
    int local = wbase + s * 32 + lane;
    int w = blockBase + local;
    unsigned bits = (w < NWORDS) ? bitmap[w] : 0u;
    s_cnt[local] = (float)__popc(bits);
  }
  __syncthreads();

  // Build operands. X[m][n] = popcount of word (16m + n), m,n in [0,16).
  // A (16x32 MxK), this code's k order k = 16*half + s (A, B match): lane half selects K+16; K>=16 = pad.
  // B (32x16 KxN), same k order as A: lane half selects K+16; K>=16 rows = pad.
  v16bf xa{}, xb{}, ub{}, la{};
#pragma unroll
  for (int s = 0; s < 16; ++s) {
    if (lhalf == 0) {
      int Ka = s;                                   // K in [0,16)
      xa[s] = (__bf16)s_cnt[wbase + 16 * ln16 + Ka];      // X as A (M=ln16)
      la[s] = (__bf16)((Ka < ln16) ? 1.0f : 0.0f);        // strict lower tri
    } else {
      xa[s] = (__bf16)0.0f;
      la[s] = (__bf16)0.0f;
    }
    if (lhalf == 0) {                               // K = s in [0,16)
      xb[s] = (__bf16)s_cnt[wbase + 16 * s + ln16];       // X as B (N=ln16)
      ub[s] = (__bf16)((s <= ln16) ? 1.0f : 0.0f);        // upper tri incl diag
    } else {                                        // K = s+16 -> zero pad
      xb[s] = (__bf16)0.0f;
      ub[s] = (__bf16)0.0f;
    }
  }

  v8f zero{};
  // R[m][n] = sum_{k<=n} X[m][k]   (row-inclusive scan)
  v8f r = __builtin_amdgcn_wmma_f32_16x16x32_bf16(false, xa, false, ub,
                                                  (short)0, zero, false, false);
  // G[m][n] = sum_{m'<m} X[m'][n]  (column partials for the row offsets)
  v8f g = __builtin_amdgcn_wmma_f32_16x16x32_bf16(false, la, false, xb,
                                                  (short)0, zero, false, false);

  // rowoff[M] = sum_n G[M][n] : reduce across the 16 lanes of each half.
  float sincl[8];
#pragma unroll
  for (int j = 0; j < 8; ++j) {
    float rs = g[j];
    rs += __shfl_xor(rs, 1, 32);
    rs += __shfl_xor(rs, 2, 32);
    rs += __shfl_xor(rs, 4, 32);
    rs += __shfl_xor(rs, 8, 32);
    sincl[j] = r[j] + rs;   // inclusive scan value at t = 16*M + N
  }

  // Wave total = inclusive value at t=255 (lane 31, vgpr slot 7).
  if (lane == 31) s_wsum[wave] = sincl[7];
  __syncthreads();
  if (tid == 0) {
    float run = 0.0f;
#pragma unroll
    for (int w2 = 0; w2 < 8; ++w2) { s_woff[w2] = run; run += s_wsum[w2]; }
  }
  __syncthreads();
  const float woff = s_woff[wave];

  // Write exclusive-within-block prefix for each word.
#pragma unroll
  for (int j = 0; j < 8; ++j) {
    int M = j + 8 * lhalf;
    int t = 16 * M + ln16;
    int w = blockBase + wbase + t;
    if (w < NWORDS)
      wordScan[w] = (int)(sincl[j] - s_cnt[wbase + t] + woff);
  }
  if (tid == 0)
    blockSums[blockIdx.x] = (unsigned)(s_woff[7] + s_wsum[7]);
}

// ---------------------------------------------------------------------------
// Single-block exclusive scan of the (<=512) block sums.
__global__ __launch_bounds__(512) void scan_top_kernel(unsigned* __restrict__ bs,
                                                       int nblk)
{
  __shared__ int s[512];
  int t = threadIdx.x;
  int v = (t < nblk) ? (int)bs[t] : 0;
  s[t] = v;
  __syncthreads();
  for (int off = 1; off < 512; off <<= 1) {
    int add = (t >= off) ? s[t - off] : 0;
    __syncthreads();
    s[t] += add;
    __syncthreads();
  }
  if (t < nblk) bs[t] = (unsigned)(s[t] - v);   // exclusive
}

// ---------------------------------------------------------------------------
// rank(key) = blockOffset + wordOffset + popcount of lower bits in word.
__global__ void rank_kernel(const int* __restrict__ si,
                            const int* __restrict__ ko,
                            int N, int K, long long NK,
                            const unsigned* __restrict__ bitmap,
                            const int*      __restrict__ wordScan,
                            const unsigned* __restrict__ blockSums,
                            int* __restrict__ out)
{
  long long i = (long long)blockIdx.x * blockDim.x + threadIdx.x;
  if (i >= NK) return;
  int n = (int)(i / K);
  int k = (int)(i % K);

  int b = si[(long long)0 * N + n];
  int x = si[(long long)1 * N + n] + ko[3 * k + 0] + PAD;
  int y = si[(long long)2 * N + n] + ko[3 * k + 1] + PAD;
  int z = si[(long long)3 * N + n] + ko[3 * k + 2] + PAD;

  int key = ((b * BASE + x) * BASE + y) * BASE + z;
  int w   = key >> 5;
  int bit = key & 31;
  unsigned below = bitmap[w] & ((1u << bit) - 1u);
  int rank = (int)blockSums[w / WPB] + wordScan[w] + __popc(below);
  out[3 * i + 1] = rank;
}

// ---------------------------------------------------------------------------
extern "C" void kernel_launch(void* const* d_in, const int* in_sizes, int n_in,
                              void* d_out, int out_size, void* d_ws, size_t ws_size,
                              hipStream_t stream) {
  const int* si = (const int*)d_in[0];        // sparse_indices [4, N]
  const int* ko = (const int*)d_in[1];        // kernel_offsets [K, 3]
  const int  N  = in_sizes[0] / 4;
  const int  K  = in_sizes[1] / 3;
  const long long NK = (long long)N * K;

  unsigned* bitmap    = (unsigned*)d_ws;                                      // NWORDS u32
  int*      wordScan  = (int*)((char*)d_ws + sizeof(unsigned) * (size_t)NWORDS);
  unsigned* blockSums = (unsigned*)((char*)d_ws + sizeof(unsigned) * (size_t)NWORDS * 2);
  int*      out       = (int*)d_out;

  zero_bitmap_kernel<<<(NWORDS + 255) / 256, 256, 0, stream>>>(bitmap, NWORDS);

  unsigned gridNK = (unsigned)((NK + 255) / 256);
  expand_kernel<<<gridNK, 256, 0, stream>>>(si, ko, N, K, NK, out, bitmap);

  scan_blocks_kernel<<<NBLK, 256, 0, stream>>>(bitmap, wordScan, blockSums);
  scan_top_kernel<<<1, 512, 0, stream>>>(blockSums, NBLK);

  rank_kernel<<<gridNK, 256, 0, stream>>>(si, ko, N, K, NK,
                                          bitmap, wordScan, blockSums, out);
}
